// Set2Set_18133351924444
// MI455X (gfx1250) — matrix operand order repaired and checked by reading, not compiled, not run on hardware
//
#include <hip/hip_runtime.h>

// ---------------------------------------------------------------------------
// Set2Set fused kernel for MI455X (gfx1250, wave32, WMMA 16x16x32 f16).
//
// Design: batch[] is sorted -> each block owns 16 consecutive graphs and their
// contiguous node range. x rows for the group are cached ONCE in LDS as f16
// (HBM floor: one 256MB read of x ~ 11us @ 23.3 TB/s). All T=6 steps run from
// LDS with zero global sync. LSTM gates and the attention-weighted pooling are
// both mapped onto v_wmma_f32_16x16x32_f16.
// ---------------------------------------------------------------------------

typedef _Float16 f16;
typedef __attribute__((ext_vector_type(16))) _Float16 v16h;
typedef __attribute__((ext_vector_type(2)))  _Float16 h2v;
typedef __attribute__((ext_vector_type(8)))  float    v8f;

#define CC       64       // channels
#define GPB      16       // graphs per block (WMMA M dimension)
#define NTH      512      // threads per block = 16 waves
#define NWAVES   16
#define NODE_CAP 1280     // LDS-cached nodes per block (mean ~976, sigma ~31)
#define XSTR     66       // f16 per cached x row (64 + 2 pad -> conflict-free)
#define KDIM     192      // LSTM GEMM K = 2C (q_star) + C (h)
#define GDIM     256      // 4*C gate outputs
#define TSTEPS   6

// ---- shared memory layout (bytes) -----------------------------------------
constexpr int SM_WCS   = 0;                        // f16[192*256] swizzled W
constexpr int SM_BIAS  = SM_WCS + KDIM * GDIM * 2; // float[256]
constexpr int SM_XS    = SM_BIAS + GDIM * 4;       // f16[NODE_CAP*XSTR]
constexpr int SM_EA    = SM_XS + NODE_CAP * XSTR * 2; // float[NODE_CAP] e/ex
constexpr int SM_GID   = SM_EA + NODE_CAP * 4;     // u8[NODE_CAP]
constexpr int SM_HS    = SM_GID + NODE_CAP;        // float[16*64]
constexpr int SM_CS    = SM_HS + GPB * CC * 4;     // float[16*64]
constexpr int SM_RS    = SM_CS + GPB * CC * 4;     // float[16*64]
constexpr int SM_GATES = SM_RS + GPB * CC * 4;     // float[16*256]
constexpr int SM_MSU   = SM_GATES + GPB * GDIM * 4;// u32[16] ordered max
constexpr int SM_ZS    = SM_MSU + GPB * 4;         // float[16] softmax denom
constexpr int SM_NN    = SM_ZS + GPB * 4;          // int[2] node range
constexpr int SM_TOTAL = SM_NN + 16;               // ~303.5 KB (< 320 KB/WG)
static_assert(SM_TOTAL <= 320 * 1024, "exceeds WGP LDS");

// order-preserving float <-> u32 (for integer atomicMax on LDS)
__device__ __forceinline__ unsigned f2ord(float f) {
  unsigned b = __float_as_uint(f);
  return (b & 0x80000000u) ? ~b : (b | 0x80000000u);
}
__device__ __forceinline__ float ord2f(unsigned u) {
  unsigned b = (u & 0x80000000u) ? (u & 0x7fffffffu) : ~u;
  return __uint_as_float(b);
}

__device__ __forceinline__ int lower_bound(const int* a, int n, int key) {
  int lo = 0, hi = n;
  while (lo < hi) {
    int mid = (lo + hi) >> 1;
    if (a[mid] < key) lo = mid + 1; else hi = mid;
  }
  return lo;
}

// ---------------------------------------------------------------------------
// Prep: pack Wc[k][n] = (k<128 ? W_ih[n][k] : W_hh[n][k-128]) as f16 in the
// exact 32x16 B-operand register layout per 32x16 tile, so the main kernel's
// lanes load operands with contiguous 32B reads.
// B order used here (32x16 KxN, contiguous k as in the A loads): lanes 0-15 hold K=0..15 (j=K), lanes 16-31
// hold K=16..31 (j=K-16); N = lane%16.
// ---------------------------------------------------------------------------
__global__ void set2set_prep(const float* __restrict__ W_ih,
                             const float* __restrict__ W_hh,
                             const float* __restrict__ b_ih,
                             const float* __restrict__ b_hh,
                             f16* __restrict__ wcw, float* __restrict__ biasG) {
  int idx = blockIdx.x * 256 + threadIdx.x;
  if (idx < KDIM * GDIM) {
    int k = idx >> 8, n = idx & 255;
    float v = (k < 128) ? W_ih[n * 128 + k] : W_hh[n * 64 + (k - 128)];
    int kt = k >> 5, kk = k & 31;
    int nt = n >> 4, nl = n & 15;
    int lanep = nl + ((kk < 16) ? 0 : 16);
    int j = kk & 15;
    wcw[((kt * 16 + nt) * 32 + lanep) * 16 + j] = (f16)v;
  }
  int bidx = idx - KDIM * GDIM;
  if (bidx >= 0 && bidx < GDIM) biasG[bidx] = b_ih[bidx] + b_hh[bidx];
}

// ---------------------------------------------------------------------------
// Main fused kernel: one block = 16 graphs, 6 steps, everything from LDS.
// ---------------------------------------------------------------------------
__global__ __launch_bounds__(NTH, 1)
void set2set_fused(const float* __restrict__ x, const int* __restrict__ batch,
                   const f16* __restrict__ wcw, const float* __restrict__ biasG,
                   float* __restrict__ out, int N) {
  extern __shared__ __align__(32) unsigned char smem[];
  f16*           wcs   = (f16*)(smem + SM_WCS);
  float*         biasS = (float*)(smem + SM_BIAS);
  f16*           xs    = (f16*)(smem + SM_XS);
  float*         ea    = (float*)(smem + SM_EA);
  unsigned char* gidA  = (unsigned char*)(smem + SM_GID);
  float*         hs    = (float*)(smem + SM_HS);
  float*         cs    = (float*)(smem + SM_CS);
  float*         rs    = (float*)(smem + SM_RS);
  float*         gates = (float*)(smem + SM_GATES);
  unsigned*      msu   = (unsigned*)(smem + SM_MSU);
  float*         zs    = (float*)(smem + SM_ZS);
  int*           nn    = (int*)(smem + SM_NN);

  const int tid  = threadIdx.x;
  const int lane = tid & 31;
  const int wave = tid >> 5;
  const int g0   = blockIdx.x * GPB;

  // stage weights + bias into LDS (shared across all 6 steps)
  for (int i = tid; i < KDIM * GDIM / 2; i += NTH)
    ((unsigned*)wcs)[i] = ((const unsigned*)wcw)[i];
  for (int i = tid; i < GDIM; i += NTH) biasS[i] = biasG[i];

  if (tid == 0) {
    nn[0] = lower_bound(batch, N, g0);
    nn[1] = lower_bound(batch, N, g0 + GPB);
  }
  for (int i = tid; i < GPB * CC; i += NTH) { hs[i] = 0.f; cs[i] = 0.f; rs[i] = 0.f; }
  __syncthreads();

  const int n0 = nn[0], n1 = nn[1];
  const int cnt = n1 - n0;
  const int cntC = (cnt < NODE_CAP) ? cnt : NODE_CAP;

  // cache x rows for this group in LDS as f16 (quarter-row per iteration)
  for (int u = tid; u < cntC * 4; u += NTH) {
    int li = u >> 2, q = u & 3;
    const float* src = x + (size_t)(n0 + li) * CC + q * 16;
    f16* dst = xs + li * XSTR + q * 16;
#pragma unroll
    for (int c = 0; c < 16; c += 2) {
      unsigned short lo = __builtin_bit_cast(unsigned short, (f16)src[c]);
      unsigned short hi = __builtin_bit_cast(unsigned short, (f16)src[c + 1]);
      *(unsigned*)(dst + c) = ((unsigned)hi << 16) | lo;
    }
    if (q == 0) gidA[li] = (unsigned char)(batch[n0 + li] - g0);
  }
  // (first in-step __syncthreads covers visibility before xs/gid are used)

  for (int step = 0; step < TSTEPS; ++step) {
    if (tid < GPB) { msu[tid] = 0u; zs[tid] = 0.f; }

    // ---- LSTM gates via WMMA: gates[16][256] = [h,r,h][16][192] @ Wc ------
    {
      const int nt = wave;            // 16 waves <-> 16 N-tiles
      const int m  = lane & 15;
      const int n  = (nt << 4) + (lane & 15);
      v8f acc = {0.f, 0.f, 0.f, 0.f, 0.f, 0.f, 0.f, 0.f};
#pragma unroll
      for (int kt = 0; kt < KDIM / 32; ++kt) {
        // A-operand (16x32 f16): lane<16 holds K {0..15}, lane>=16
        // holds K {16..31}; row M = lane%16.
        v16h a;
#pragma unroll
        for (int j = 0; j < 16; ++j) {
          int k = kt * 32 + j + ((lane < 16) ? 0 : 16);
          float v;
          if (k < 64)       v = hs[m * CC + k];          // q_prev = h_prev
          else if (k < 128) v = rs[m * CC + (k - 64)];   // r_prev
          else              v = hs[m * CC + (k - 128)];  // h_prev (W_hh part)
          a[j] = (f16)v;
        }
        v16h b = *(const v16h*)(wcs + ((kt * 16 + nt) * 32 + lane) * 16);
        acc = __builtin_amdgcn_wmma_f32_16x16x32_f16(
            false, a, false, b, (short)0, acc, false, false);
      }
      float bv = biasS[n];
#pragma unroll
      for (int v = 0; v < 8; ++v) {
        int mm = v + ((lane < 16) ? 0 : 8);   // C/D layout: M = v (+8 hi half)
        gates[mm * GDIM + n] = acc[v] + bv;
      }
    }
    __syncthreads();

    // ---- activations: c = sig(f)*c + sig(i)*tanh(g); h = sig(o)*tanh(c) ---
    for (int p = tid; p < GPB * CC; p += NTH) {
      int m = p >> 6, jc = p & 63;
      float gi = gates[m * GDIM + jc];
      float gf = gates[m * GDIM + 64 + jc];
      float gg = gates[m * GDIM + 128 + jc];
      float go = gates[m * GDIM + 192 + jc];
      float si = 1.f / (1.f + __expf(-gi));
      float sf = 1.f / (1.f + __expf(-gf));
      float so = 1.f / (1.f + __expf(-go));
      float cn = sf * cs[p] + si * tanhf(gg);
      cs[p] = cn;
      hs[p] = so * tanhf(cn);   // h == q for the attention pass
      rs[p] = 0.f;              // clear r accumulator for this step
    }
    __syncthreads();

    // ---- e[n] = x[n] . q[gid(n)]  + per-graph max (ordered-u32 atomicMax) -
    for (int li = tid; li < cntC; li += NTH) {
      int g = gidA[li];
      const f16* xr = xs + li * XSTR;
      const float* q = hs + g * CC;
      float e = 0.f;
#pragma unroll
      for (int c = 0; c < CC; c += 2) {
        h2v hh = __builtin_bit_cast(h2v, *(const unsigned*)(xr + c));
        e += (float)hh.x * q[c] + (float)hh.y * q[c + 1];
      }
      ea[li] = e;
      atomicMax(&msu[g], f2ord(e));
    }
    for (int li = NODE_CAP + tid; li < cnt; li += NTH) {  // overflow fallback
      int g = batch[n0 + li] - g0;
      const float* xr = x + (size_t)(n0 + li) * CC;
      float e = 0.f;
      for (int c = 0; c < CC; ++c) e += xr[c] * hs[g * CC + c];
      atomicMax(&msu[g], f2ord(e));
    }
    __syncthreads();

    // ---- ex = exp(e - m),  z = segment_sum(ex) ---------------------------
    for (int li = tid; li < cntC; li += NTH) {
      int g = gidA[li];
      float ex = __expf(ea[li] - ord2f(msu[g]));
      ea[li] = ex;
      atomicAdd(&zs[g], ex);
    }
    for (int li = NODE_CAP + tid; li < cnt; li += NTH) {  // overflow fallback
      int g = batch[n0 + li] - g0;
      const float* xr = x + (size_t)(n0 + li) * CC;
      float e = 0.f;
      for (int c = 0; c < CC; ++c) e += xr[c] * hs[g * CC + c];
      float ex = __expf(e - ord2f(msu[g]));
      atomicAdd(&zs[g], ex);
      for (int c = 0; c < CC; ++c) atomicAdd(&rs[g * CC + c], ex * xr[c]);
    }
    __syncthreads();

    // ---- r[m][:] += sum_k ex[k]*[gid==m] * x[k][:] via WMMA over chunks ---
    {
      const int m  = lane & 15;
      const int nb = lane & 15;
      v8f acc[4];
#pragma unroll
      for (int nt = 0; nt < 4; ++nt)
        acc[nt] = (v8f){0.f, 0.f, 0.f, 0.f, 0.f, 0.f, 0.f, 0.f};
      const int nch = (cntC + 31) >> 5;
      for (int ch = wave; ch < nch; ch += NWAVES) {
        v16h a;                                  // masked ex -> A[16x32]
#pragma unroll
        for (int j = 0; j < 16; ++j) {
          int k  = j + ((lane < 16) ? 0 : 16);
          int li = (ch << 5) + k;
          float v = 0.f;
          if (li < cntC && gidA[li] == m) v = ea[li];
          a[j] = (f16)v;
        }
#pragma unroll
        for (int nt = 0; nt < 4; ++nt) {         // B = x rows [32 x 16ch]
          v16h b;
#pragma unroll
          for (int j = 0; j < 16; ++j) {
            int li = (ch << 5) + ((lane < 16) ? 0 : 16) + j;
            f16 v = (f16)0.f;
            if (li < cntC) v = xs[li * XSTR + (nt << 4) + nb];
            b[j] = v;
          }
          acc[nt] = __builtin_amdgcn_wmma_f32_16x16x32_f16(
              false, a, false, b, (short)0, acc[nt], false, false);
        }
      }
#pragma unroll
      for (int nt = 0; nt < 4; ++nt)
#pragma unroll
        for (int v = 0; v < 8; ++v) {
          int mm = v + ((lane < 16) ? 0 : 8);
          atomicAdd(&rs[mm * CC + (nt << 4) + nb], acc[nt][v]);
        }
    }
    __syncthreads();

    // ---- deferred softmax normalization: r /= z --------------------------
    for (int p = tid; p < GPB * CC; p += NTH) {
      float z = zs[p >> 6];
      rs[p] = (z > 0.f) ? rs[p] / z : 0.f;
    }
    __syncthreads();
  }

  // q_star = [h, r]
  for (int p = tid; p < GPB * 2 * CC; p += NTH) {
    int m = p >> 7, j = p & 127;
    out[(size_t)(g0 + m) * (2 * CC) + j] = (j < CC) ? hs[m * CC + j]
                                                    : rs[m * CC + (j - CC)];
  }
}

// ---------------------------------------------------------------------------
extern "C" void kernel_launch(void* const* d_in, const int* in_sizes, int n_in,
                              void* d_out, int out_size, void* d_ws,
                              size_t ws_size, hipStream_t stream) {
  const float* x     = (const float*)d_in[0];
  const int*   batch = (const int*)d_in[1];
  const float* W_ih  = (const float*)d_in[2];
  const float* W_hh  = (const float*)d_in[3];
  const float* b_ih  = (const float*)d_in[4];
  const float* b_hh  = (const float*)d_in[5];
  float*       out   = (float*)d_out;

  const int N  = in_sizes[1];          // node count (batch[] length)
  const int Bv = out_size / (2 * CC);  // graph count

  // ws: [0,96KB) swizzled f16 weights, then 1KB combined bias
  f16*   wcw   = (f16*)d_ws;
  float* biasG = (float*)((char*)d_ws + KDIM * GDIM * 2);

  (void)n_in; (void)ws_size;
  (void)hipFuncSetAttribute((const void*)set2set_fused,
                            hipFuncAttributeMaxDynamicSharedMemorySize,
                            SM_TOTAL);

  set2set_prep<<<(KDIM * GDIM + GDIM + 255) / 256, 256, 0, stream>>>(
      W_ih, W_hh, b_ih, b_hh, wcw, biasG);

  set2set_fused<<<Bv / GPB, NTH, SM_TOTAL, stream>>>(x, batch, wcw, biasG, out, N);
}
